// LinearPerformerAttention_6717328851263
// MI455X (gfx1250) — matrix operand order repaired and checked by reading, not compiled, not run on hardware
//
#include <hip/hip_runtime.h>
#include <hip/hip_bf16.h>

// ---------------------------------------------------------------------------
// Linear Performer attention for MI455X (gfx1250), wave32 + WMMA bf16.
// All GEMMs consume B pre-transposed (BT[N][K]) so LDS staging is pure b128.
// Double-buffered LDS; gfx1250 async global->LDS staging (ASYNCcnt).
// ---------------------------------------------------------------------------

typedef __bf16 bf16_t;
typedef __attribute__((ext_vector_type(16))) __bf16 v16bf;
typedef __attribute__((ext_vector_type(8)))  __bf16 v8bf;
typedef __attribute__((ext_vector_type(8)))  float  v8f;
typedef __attribute__((ext_vector_type(4)))  int    v4i;

#define EPI_BF16        0
#define EPI_ELU_BF16    1
#define EPI_ELU_BF16_T  2
#define EPI_ZSCALE_BF16 3
#define EPI_BIAS_F32    4
#define EPI_BF16_T      5

#if defined(__has_builtin)
#if __has_builtin(__builtin_amdgcn_global_load_async_to_lds_b128)
#define HAVE_ASYNC_LDS 1
#endif
#endif
#ifndef HAVE_ASYNC_LDS
#define HAVE_ASYNC_LDS 0
#endif

#if HAVE_ASYNC_LDS
#define ASYNC_GPTR(p) ((__attribute__((address_space(1))) v4i*)(p))
#define ASYNC_LPTR(p) ((__attribute__((address_space(3))) v4i*)(p))
#endif

struct GemmParams {
    const bf16_t* A;      // [M x K], leading dim lda
    const bf16_t* B;      // BT layout: [N x K], leading dim ldb (= K stride)
    void*         C;      // bf16 or f32 depending on epilogue
    int  K, lda, ldb, ldc;
    long aOuter, aInner;  // batch strides, bz = 16*outer_idx + inner_idx
    long bOuter, bInner;
    long cOuter, cInner;
    const float* aux;     // z vector or bias
    long auxStride;       // per-batch stride for aux (z)
};

// ---------------------------------------------------------------------------
// fp32 -> bf16 elementwise convert
// ---------------------------------------------------------------------------
__global__ __launch_bounds__(256) void convert_f32_bf16(const float* __restrict__ in,
                                                        bf16_t* __restrict__ out, long n) {
    long i = (long)blockIdx.x * blockDim.x + threadIdx.x;
    long stride = (long)gridDim.x * blockDim.x;
    for (; i < n; i += stride) out[i] = (bf16_t)in[i];
}

// ---------------------------------------------------------------------------
// Tiled 32x32 transpose (+convert) into bf16. Block = 32x8 threads.
// out[c][r] = in[r][c]. Batched via bz = 16*bo + bi stride decomposition.
// ---------------------------------------------------------------------------
template <typename TIN>
__global__ __launch_bounds__(256) void transpose_to_bf16(
    const TIN* __restrict__ in, bf16_t* __restrict__ out, int ldi, int ldo,
    long inOuter, long inInner, long outOuter, long outInner) {
    __shared__ bf16_t tile[32][33];
    const long bz = blockIdx.z, bo = bz >> 4, bi = bz & 15;
    const TIN* ib = in + bo * inOuter + bi * inInner;
    bf16_t* ob = out + bo * outOuter + bi * outInner;
    const int tx = threadIdx.x, ty = threadIdx.y;
    const int r0 = blockIdx.y * 32, c0 = blockIdx.x * 32;
#pragma unroll
    for (int j = 0; j < 4; ++j)
        tile[ty + j * 8][tx] = (bf16_t)ib[(long)(r0 + ty + j * 8) * ldi + c0 + tx];
    __syncthreads();
#pragma unroll
    for (int j = 0; j < 4; ++j)
        ob[(long)(c0 + ty + j * 8) * ldo + r0 + tx] = tile[tx][ty + j * 8];
}

// ---------------------------------------------------------------------------
// Generic bf16 WMMA GEMM, B pre-transposed. 256 threads = 8 waves (wave32).
// BK=32, double-buffered LDS. M%BM==0, N%BN==0, K%32==0.
// ---------------------------------------------------------------------------
template <int BM, int BN, int WMC, int WNC, int EPI>
__global__ __launch_bounds__(256) void gemm_bf16_wmma(GemmParams p) {
    constexpr int WM = BM / WMC;
    constexpr int WN = BN / WNC;
    constexpr int TM = WM / 16;
    constexpr int TN = WN / 16;
    constexpr int LDP = 40;  // padded LDS stride (conflict-free b128)

    __shared__ bf16_t As[2][BM * LDP];  // [BM][32] row-major (padded)
    __shared__ bf16_t Bs[2][BN * LDP];  // [BN][32] (BT rows, padded)

    const int tid  = threadIdx.x;
    const int wid  = tid >> 5;
    const int lane = tid & 31;
    const int waveM = wid / WNC;
    const int waveN = wid % WNC;

    const long bz = blockIdx.z;
    const long bo = bz >> 4, bi = bz & 15;
    const bf16_t* Ab = p.A + bo * p.aOuter + bi * p.aInner;
    const bf16_t* Bb = p.B + bo * p.bOuter + bi * p.bInner;
    const long cBase = bo * p.cOuter + bi * p.cInner;

    const int blockM = blockIdx.y * BM;
    const int blockN = blockIdx.x * BN;

    const v8f vzero = {0.f, 0.f, 0.f, 0.f, 0.f, 0.f, 0.f, 0.f};
    v8f acc[TM][TN];
#pragma unroll
    for (int tm = 0; tm < TM; ++tm)
#pragma unroll
        for (int tn = 0; tn < TN; ++tn) acc[tm][tn] = vzero;

    constexpr int A_CH = (BM * 32) / (256 * 8);  // b128 units per thread
    constexpr int B_CH = (BN * 32) / (256 * 8);

    const int mlocal = lane & 15;
    const int ksel   = lane >> 4;

    // stage one K-slab (32 wide) of A and BT into LDS buffer `buf`
    auto stage = [&](int buf, int kk) {
#pragma unroll
        for (int c = 0; c < A_CH; ++c) {
            int idx  = tid + c * 256;
            int row  = idx >> 2;
            int colk = (idx & 3) * 8;
            const bf16_t* g = Ab + (long)(blockM + row) * p.lda + kk + colk;
            bf16_t* l = &As[buf][row * LDP + colk];
#if HAVE_ASYNC_LDS
            __builtin_amdgcn_global_load_async_to_lds_b128(ASYNC_GPTR(g), ASYNC_LPTR(l), 0, 0);
#else
            *reinterpret_cast<v8bf*>(l) = *reinterpret_cast<const v8bf*>(g);
#endif
        }
#pragma unroll
        for (int c = 0; c < B_CH; ++c) {
            int idx  = tid + c * 256;
            int row  = idx >> 2;
            int colk = (idx & 3) * 8;
            const bf16_t* g = Bb + (long)(blockN + row) * p.ldb + kk + colk;
            bf16_t* l = &Bs[buf][row * LDP + colk];
#if HAVE_ASYNC_LDS
            __builtin_amdgcn_global_load_async_to_lds_b128(ASYNC_GPTR(g), ASYNC_LPTR(l), 0, 0);
#else
            *reinterpret_cast<v8bf*>(l) = *reinterpret_cast<const v8bf*>(g);
#endif
        }
    };

    const int nk = p.K >> 5;
    stage(0, 0);

    for (int it = 0; it < nk; ++it) {
        const int cur = it & 1;
#if HAVE_ASYNC_LDS
        asm volatile("s_wait_asynccnt 0x0" ::: "memory");
#endif
        __syncthreads();
        if (it + 1 < nk) stage((it + 1) & 1, (it + 1) * 32);

        // ---- fragments (ISA 7.12.2 16-bit A/B wave32 layouts) ----
        union Frag { v16bf v; v8bf h[2]; };
        Frag afrag[TM];
#pragma unroll
        for (int tm = 0; tm < TM; ++tm) {
            int m  = waveM * WM + tm * 16 + mlocal;
            int kh = ksel * 8;  // lanes 0-15: K 0..7/16..23 ; lanes 16-31: K 8..15/24..31
            afrag[tm].h[0] = *reinterpret_cast<const v8bf*>(&As[cur][m * LDP + kh]);
            afrag[tm].h[1] = *reinterpret_cast<const v8bf*>(&As[cur][m * LDP + kh + 16]);
        }
        Frag bfrag[TN];
#pragma unroll
        for (int tn = 0; tn < TN; ++tn) {
            int n  = waveN * WN + tn * 16 + mlocal;
            int k0 = ksel * 8;  // lanes 0-15: K 0..7/16..23 ; lanes 16-31: K 8..15/24..31
            bfrag[tn].h[0] = *reinterpret_cast<const v8bf*>(&Bs[cur][n * LDP + k0]);
            bfrag[tn].h[1] = *reinterpret_cast<const v8bf*>(&Bs[cur][n * LDP + k0 + 16]);
        }

#pragma unroll
        for (int tm = 0; tm < TM; ++tm)
#pragma unroll
            for (int tn = 0; tn < TN; ++tn)
                acc[tm][tn] = __builtin_amdgcn_wmma_f32_16x16x32_bf16(
                    false, afrag[tm].v, false, bfrag[tn].v,
                    (short)0, acc[tm][tn], false, false);
    }

    // ---- epilogue (C layout: lane<16 -> M=r, lane>=16 -> M=8+r; N=lane&15) ----
    const int rowHalf = ksel * 8;
#pragma unroll
    for (int tm = 0; tm < TM; ++tm) {
#pragma unroll
        for (int tn = 0; tn < TN; ++tn) {
#pragma unroll
            for (int r = 0; r < 8; ++r) {
                int row = blockM + waveM * WM + tm * 16 + rowHalf + r;
                int col = blockN + waveN * WN + tn * 16 + mlocal;
                float v = acc[tm][tn][r];
                if constexpr (EPI == EPI_ELU_BF16 || EPI == EPI_ELU_BF16_T)
                    v = (v > 0.f) ? (v + 1.f) : __expf(v);
                if constexpr (EPI == EPI_ZSCALE_BF16)
                    v *= p.aux[bz * p.auxStride + row];

                if constexpr (EPI == EPI_BIAS_F32) {
                    ((float*)p.C)[cBase + (long)row * p.ldc + col] = v + p.aux[col];
                } else if constexpr (EPI == EPI_ELU_BF16_T || EPI == EPI_BF16_T) {
                    ((bf16_t*)p.C)[cBase + (long)col * p.ldc + row] = (bf16_t)v;
                } else {
                    ((bf16_t*)p.C)[cBase + (long)row * p.ldc + col] = (bf16_t)v;
                }
            }
        }
    }
}

// ---------------------------------------------------------------------------
// ksum[bh*256+f] = sum_n KpT[bh][f][n]   (one block per (bh,f), n contiguous)
// ---------------------------------------------------------------------------
__global__ __launch_bounds__(128) void ksum_kernel(const bf16_t* __restrict__ KpT,
                                                   float* __restrict__ ksum) {
    __shared__ float red[128];
    const long blk = blockIdx.x;  // bh*256 + f
    const bf16_t* src = KpT + blk * 4096;
    float s = 0.f;
    for (int i = threadIdx.x; i < 4096; i += 128) s += (float)src[i];
    red[threadIdx.x] = s;
    __syncthreads();
    for (int off = 64; off > 0; off >>= 1) {
        if (threadIdx.x < off) red[threadIdx.x] += red[threadIdx.x + off];
        __syncthreads();
    }
    if (threadIdx.x == 0) ksum[blk] = red[0];
}

// ---------------------------------------------------------------------------
// z[bh][n] = 1/(dot(Qp[bh][n][:], ksum[bh][:]) + 1e-8). One wave per row.
// ---------------------------------------------------------------------------
__global__ __launch_bounds__(256) void z_kernel(const bf16_t* __restrict__ Qp,
                                                const float* __restrict__ ksum,
                                                float* __restrict__ z) {
    const int wid  = threadIdx.x >> 5;
    const int lane = threadIdx.x & 31;
    const long row = (long)blockIdx.x * 8 + wid;  // 64*4096 rows total
    const long bh  = row >> 12;
    const bf16_t* q = Qp + row * 256 + lane * 8;
    const float*  k = ksum + bh * 256 + lane * 8;
    v8bf qv = *reinterpret_cast<const v8bf*>(q);
    float s = 0.f;
#pragma unroll
    for (int j = 0; j < 8; ++j) s += (float)qv[j] * k[j];
    for (int off = 16; off > 0; off >>= 1) s += __shfl_xor(s, off, 32);
    if (lane == 0) z[row] = 1.f / (s + 1e-8f);
}

// ---------------------------------------------------------------------------
// Host launch
// ---------------------------------------------------------------------------
extern "C" void kernel_launch(void* const* d_in, const int* in_sizes, int n_in,
                              void* d_out, int out_size, void* d_ws, size_t ws_size,
                              hipStream_t stream) {
    const float* x    = (const float*)d_in[0];  // (4,4096,1024)
    const float* wqkv = (const float*)d_in[1];  // (1024,3072)
    const float* proj = (const float*)d_in[2];  // (16,64,256)
    const float* wout = (const float*)d_in[3];  // (1024,1024)
    const float* bout = (const float*)d_in[4];  // (1024,)
    float* out = (float*)d_out;

    const long M = 4L * 4096;  // 16384 tokens
    const long D = 1024, D3 = 3072, HD = 64, F = 256, NT = 4096;
    const long BH = 64;

    // ---- workspace layout ----
    char* w = (char*)d_ws;
    auto alloc = [&](size_t bytes) {
        void* r = (void*)w;
        w += (bytes + 255) & ~(size_t)255;
        return r;
    };
    bf16_t* Xb     = (bf16_t*)alloc(M * D * 2);
    bf16_t* WqkvT  = (bf16_t*)alloc(D * D3 * 2);       // [3072][1024]
    bf16_t* PT     = (bf16_t*)alloc(16 * F * HD * 2);  // [h][256][64]
    bf16_t* WoutT  = (bf16_t*)alloc(D * D * 2);        // [1024][1024]
    bf16_t* QKVb   = (bf16_t*)alloc(M * D3 * 2);
    bf16_t* VT     = (bf16_t*)alloc(BH * HD * NT * 2); // [bh][64][4096]
    bf16_t* Qp     = (bf16_t*)alloc(BH * NT * F * 2);  // [bh][n][f]
    bf16_t* KpT    = (bf16_t*)alloc(BH * F * NT * 2);  // [bh][f][n]
    float*  ksum   = (float*)alloc(BH * F * 4);
    bf16_t* KVT    = (bf16_t*)alloc(BH * HD * F * 2);  // [bh][d][f]
    float*  zbuf   = (float*)alloc(BH * NT * 4);
    bf16_t* Amerge = (bf16_t*)alloc(M * D * 2);
    (void)ws_size; (void)in_sizes; (void)n_in; (void)out_size;

    // ---- 0) converts / weight transposes ----
    convert_f32_bf16<<<1024, 256, 0, stream>>>(x, Xb, M * D);
    transpose_to_bf16<float><<<dim3(96, 32, 1), dim3(32, 8), 0, stream>>>(
        wqkv, WqkvT, 3072, 1024, 0, 0, 0, 0);
    transpose_to_bf16<float><<<dim3(8, 2, 16), dim3(32, 8), 0, stream>>>(
        proj, PT, 256, 64, 0, HD * F, 0, F * HD);
    transpose_to_bf16<float><<<dim3(32, 32, 1), dim3(32, 8), 0, stream>>>(
        wout, WoutT, 1024, 1024, 0, 0, 0, 0);

    // ---- 1) QKV = Xb @ Wqkv  (16384 x 1024 x 3072), bf16 out ----
    {
        GemmParams p{};
        p.A = Xb; p.B = WqkvT; p.C = QKVb;
        p.K = 1024; p.lda = 1024; p.ldb = 1024; p.ldc = 3072;
        gemm_bf16_wmma<128, 128, 2, 4, EPI_BF16>
            <<<dim3(24, 128, 1), 256, 0, stream>>>(p);
    }
    // ---- 1b) VT[bh][d][n] = transpose of V region ----
    transpose_to_bf16<bf16_t><<<dim3(2, 128, 64), dim3(32, 8), 0, stream>>>(
        QKVb + 2048, VT, 3072, 4096, NT * D3, HD, 16 * HD * NT, HD * NT);

    // ---- 2) feature projection, batched over (b,h): (4096 x 64 x 256) ----
    {
        GemmParams p{};
        p.A = QKVb; p.B = PT; p.C = Qp;
        p.K = 64; p.lda = 3072; p.ldb = 64; p.ldc = 256;
        p.aOuter = NT * D3; p.aInner = HD;   // b, h offsets into QKV (q region)
        p.bOuter = 0;       p.bInner = F * HD;
        p.cInner = NT * F;  p.cOuter = 16 * p.cInner;
        gemm_bf16_wmma<128, 128, 2, 4, EPI_ELU_BF16>
            <<<dim3(2, 32, BH), 256, 0, stream>>>(p);

        p.A = QKVb + 1024;  // k region
        p.C = KpT; p.ldc = 4096;  // transposed store: KpT[bh][f][n]
        p.cInner = F * NT; p.cOuter = 16 * p.cInner;
        gemm_bf16_wmma<128, 128, 2, 4, EPI_ELU_BF16_T>
            <<<dim3(2, 32, BH), 256, 0, stream>>>(p);
    }
    // ---- 3) ksum + z ----
    ksum_kernel<<<BH * F, 128, 0, stream>>>(KpT, ksum);
    z_kernel<<<(BH * NT) / 8, 256, 0, stream>>>(Qp, ksum, zbuf);

    // ---- 4) KV = KpT @ V  (256 x 4096 x 64), stored transposed -> KVT[bh][d][f] ----
    {
        GemmParams p{};
        p.A = KpT; p.B = VT; p.C = KVT;
        p.K = 4096; p.lda = 4096; p.ldb = 4096; p.ldc = 256;
        p.aInner = F * NT;  p.aOuter = 16 * p.aInner;
        p.bInner = HD * NT; p.bOuter = 16 * p.bInner;
        p.cInner = HD * F;  p.cOuter = 16 * p.cInner;
        gemm_bf16_wmma<128, 64, 4, 2, EPI_BF16_T>
            <<<dim3(1, 2, BH), 256, 0, stream>>>(p);
    }
    // ---- 5) Attn = (Qp @ KV) * z -> merged heads (4096 x 256 x 64) ----
    {
        GemmParams p{};
        p.A = Qp; p.B = KVT; p.C = Amerge;
        p.K = 256; p.lda = 256; p.ldb = 256; p.ldc = 1024;
        p.aInner = NT * F;  p.aOuter = 16 * p.aInner;
        p.bInner = HD * F;  p.bOuter = 16 * p.bInner;
        p.cInner = HD;      p.cOuter = NT * D;  // [b][n][h*64+d]
        p.aux = zbuf; p.auxStride = NT;
        gemm_bf16_wmma<128, 64, 4, 2, EPI_ZSCALE_BF16>
            <<<dim3(1, 32, BH), 256, 0, stream>>>(p);
    }
    // ---- 6) OUT = Amerge @ Wout + b_out  (16384 x 1024 x 1024), f32 ----
    {
        GemmParams p{};
        p.A = Amerge; p.B = WoutT; p.C = out;
        p.K = 1024; p.lda = 1024; p.ldb = 1024; p.ldc = 1024;
        p.aux = bout;
        gemm_bf16_wmma<128, 128, 2, 4, EPI_BIAS_F32>
            <<<dim3(8, 128, 1), 256, 0, stream>>>(p);
    }
}
